// MultiHeadAttention_47742856462676
// MI455X (gfx1250) — matrix operand order repaired and checked by reading, not compiled, not run on hardware
//
#include <hip/hip_runtime.h>
#include <stdint.h>

// ---------------------------------------------------------------------------
// MHA for MI455X (gfx1250, wave32, WMMA 16x16x32 f16).
//   B=16 heads, S=2048, D=1024, DH=64.
//   k0: one-shot fp32 -> f16 weight conversion (vectorized)
//   k1: q/k/v projections (WMMA f16 GEMM, bias + 1/sqrt(DH) folded, V stored T)
//   k2: flash attention, 4 waves/WG sharing K/V key-blocks staged in LDS via
//       global_load_async_to_lds_b128 (double-buffered, s_wait_asynccnt)
//   k3: output projection (WMMA f16 GEMM, f32 out + bias)
// ---------------------------------------------------------------------------

#define S_LEN   2048
#define D_MODEL 1024
#define NHEAD   16
#define DHEAD   64
#define KB_KEYS 32      // keys per flash iteration

typedef __attribute__((ext_vector_type(16))) _Float16 v16h;
typedef __attribute__((ext_vector_type(8)))  float    v8f;

union V16 {
    v16h     v;
    uint4    u4[2];
    _Float16 e[16];
};

__device__ __forceinline__ _Float16 f2h(float f) { return (_Float16)f; }

// ---------------------------------------------------------------------------
// Kernel 0: fp32 -> f16, 4 elements/thread (b128 in, b64 out)
// ---------------------------------------------------------------------------
__global__ void __launch_bounds__(256)
cvt_w_kernel(const float* __restrict__ src, _Float16* __restrict__ dst, int n) {
    const int i = (blockIdx.x * 256 + threadIdx.x) * 4;
    if (i < n) {
        float4 f = *(const float4*)(src + i);
        union { _Float16 e[4]; uint2 u; } o;
        o.e[0] = f2h(f.x); o.e[1] = f2h(f.y);
        o.e[2] = f2h(f.z); o.e[3] = f2h(f.w);
        *(uint2*)(dst + i) = o.u;
    }
}

// ---------------------------------------------------------------------------
// Kernel 1: projections.  grid = (S/16, B, 3), block = 32 (one wave).
//   q -> [B][S][DH] f16 scaled by 1/sqrt(DH);  k -> [B][S][DH];  v -> [B][DH][S]
// ---------------------------------------------------------------------------
__global__ void __launch_bounds__(32)
proj_qkv_kernel(const float* __restrict__ Q, const float* __restrict__ K,
                const float* __restrict__ V,
                const _Float16* __restrict__ Wq16, const float* __restrict__ bq,
                const _Float16* __restrict__ Wk16, const float* __restrict__ bk,
                const _Float16* __restrict__ Wv16, const float* __restrict__ bv,
                _Float16* __restrict__ qws, _Float16* __restrict__ kws,
                _Float16* __restrict__ vtws) {
    const int l    = threadIdx.x;
    const int half = l >> 4;
    const int lan  = l & 15;
    const int row0 = blockIdx.x * 16;
    const int b    = blockIdx.y;
    const int which = blockIdx.z;   // 0=q 1=k 2=v

    const float*    X    = (which == 0) ? Q    : (which == 1) ? K    : V;
    const _Float16* W    = (which == 0) ? Wq16 : (which == 1) ? Wk16 : Wv16;
    const float*    bias = (which == 0) ? bq   : (which == 1) ? bk   : bv;

    const float* Xrow = X + ((size_t)b * S_LEN + row0 + lan) * D_MODEL;

    v8f c[4] = {};

    for (int k0 = 0; k0 < D_MODEL; k0 += 32) {
        // A tile 16x32 (lane<16: K=0..7,16..23; lane>=16: +8), fp32 -> f16
        V16 a;
        const int ka = k0 + half * 8;
        float4 f0 = *(const float4*)(Xrow + ka);
        float4 f1 = *(const float4*)(Xrow + ka + 4);
        float4 f2 = *(const float4*)(Xrow + ka + 16);
        float4 f3 = *(const float4*)(Xrow + ka + 20);
        a.e[0]=f2h(f0.x);  a.e[1]=f2h(f0.y);  a.e[2]=f2h(f0.z);  a.e[3]=f2h(f0.w);
        a.e[4]=f2h(f1.x);  a.e[5]=f2h(f1.y);  a.e[6]=f2h(f1.z);  a.e[7]=f2h(f1.w);
        a.e[8]=f2h(f2.x);  a.e[9]=f2h(f2.y);  a.e[10]=f2h(f2.z); a.e[11]=f2h(f2.w);
        a.e[12]=f2h(f3.x); a.e[13]=f2h(f3.y); a.e[14]=f2h(f3.z); a.e[15]=f2h(f3.w);

        // B tiles 32x16 from pre-converted f16 weights: pure b128 loads
        const int kb = k0 + half * 8;
#pragma unroll
        for (int t = 0; t < 4; ++t) {
            const _Float16* wr = W + (size_t)(t * 16 + lan) * D_MODEL + kb;
            V16 bm;
            bm.u4[0] = *(const uint4*)wr;
            bm.u4[1] = *(const uint4*)(wr + 16);
            c[t] = __builtin_amdgcn_wmma_f32_16x16x32_f16(
                false, a.v, false, bm.v, (short)0, c[t], false, false);
        }
    }

    // Epilogue: which is wave-uniform -> one branch, then straight store loops.
    const float qscale = 0.125f; // 1/sqrt(64)
    if (which == 0) {
        _Float16* dst = qws + (size_t)b * S_LEN * DHEAD;
#pragma unroll
        for (int t = 0; t < 4; ++t) {
            const int col = t * 16 + lan;
            const float bval = bias[col];
#pragma unroll
            for (int r = 0; r < 8; ++r) {
                const int row = row0 + r + half * 8;
                dst[(size_t)row * DHEAD + col] = f2h((c[t][r] + bval) * qscale);
            }
        }
    } else if (which == 1) {
        _Float16* dst = kws + (size_t)b * S_LEN * DHEAD;
#pragma unroll
        for (int t = 0; t < 4; ++t) {
            const int col = t * 16 + lan;
            const float bval = bias[col];
#pragma unroll
            for (int r = 0; r < 8; ++r) {
                const int row = row0 + r + half * 8;
                dst[(size_t)row * DHEAD + col] = f2h(c[t][r] + bval);
            }
        }
    } else {
        _Float16* dst = vtws + (size_t)b * DHEAD * S_LEN;
#pragma unroll
        for (int t = 0; t < 4; ++t) {
            const int col = t * 16 + lan;
            const float bval = bias[col];
#pragma unroll
            for (int r = 0; r < 8; ++r) {
                const int row = row0 + r + half * 8;
                dst[(size_t)col * S_LEN + row] = f2h(c[t][r] + bval);
            }
        }
    }
}

// ---------------------------------------------------------------------------
// Kernel 2: flash attention.  grid = (S/64, B), block = 128 (4 waves).
// Each wave owns 16 query rows; the 4 waves share K/V key-blocks staged into
// LDS with async copies (double-buffered).
// ---------------------------------------------------------------------------
__global__ void __launch_bounds__(128)
flash_attn_kernel(const _Float16* __restrict__ qws,
                  const _Float16* __restrict__ kws,
                  const _Float16* __restrict__ vtws,
                  _Float16* __restrict__ att) {
    const int tid  = threadIdx.x;
    const int wave = tid >> 5;
    const int l    = tid & 31;
    const int half = l >> 4;
    const int lan  = l & 15;
    const int row0 = blockIdx.x * 64 + wave * 16;
    const int b    = blockIdx.y;

    __shared__ _Float16 kbuf[2][KB_KEYS * DHEAD];  // [key][dh]   4 KB each
    __shared__ _Float16 vbuf[2][DHEAD * KB_KEYS];  // [h][key]    4 KB each
    __shared__ _Float16 pbuf[4][16 * 32];          // per-wave P tile

    const _Float16* kbase = kws  + (size_t)b * S_LEN * DHEAD;
    const _Float16* vbase = vtws + (size_t)b * DHEAD * S_LEN;

    // Stage one 32-key block (4 KB of K + 4 KB of V) with 4 async b128/thread.
    auto stage = [&](int par, int key0) {
#pragma unroll
        for (int i = 0; i < 2; ++i) {            // K: 256 chunks of 16B
            const int c = tid + i * 128;
            const int key = c >> 3, part = c & 7;
            unsigned lds = (unsigned)(uintptr_t)(void*)
                &kbuf[par][key * DHEAD + part * 8];
            const void* g = (const void*)
                (kbase + (size_t)(key0 + key) * DHEAD + part * 8);
            asm volatile("global_load_async_to_lds_b128 %0, %1, off"
                         :: "v"(lds), "v"(g) : "memory");
        }
#pragma unroll
        for (int i = 0; i < 2; ++i) {            // V: 256 chunks of 16B
            const int c = tid + i * 128;
            const int h = c >> 2, part = c & 3;
            unsigned lds = (unsigned)(uintptr_t)(void*)
                &vbuf[par][h * KB_KEYS + part * 8];
            const void* g = (const void*)
                (vbase + (size_t)h * S_LEN + key0 + part * 8);
            asm volatile("global_load_async_to_lds_b128 %0, %1, off"
                         :: "v"(lds), "v"(g) : "memory");
        }
    };

    // Q A-tiles (16 x 64 = two 16x32 K-steps) resident in registers.
    V16 aq[2];
    {
        const _Float16* qrow = qws + ((size_t)b * S_LEN + row0 + lan) * DHEAD;
#pragma unroll
        for (int s = 0; s < 2; ++s) {
            const int ka = s * 32 + half * 8;
            aq[s].u4[0] = *(const uint4*)(qrow + ka);
            aq[s].u4[1] = *(const uint4*)(qrow + ka + 16);
        }
    }

    v8f acc[4] = {};
    float m[8], lsum[8];
#pragma unroll
    for (int r = 0; r < 8; ++r) { m[r] = -3.0e38f; lsum[r] = 0.0f; }

    stage(0, 0);   // prologue: fill buffer 0

    for (int key0 = 0; key0 < S_LEN; key0 += KB_KEYS) {
        const int  par  = (key0 >> 5) & 1;
        const bool more = (key0 + KB_KEYS) < S_LEN;

        if (more) {
            stage(par ^ 1, key0 + KB_KEYS);                    // prefetch next
            asm volatile("s_wait_asynccnt 0x4" ::: "memory");  // current done
        } else {
            asm volatile("s_wait_asynccnt 0x0" ::: "memory");
        }
        __syncthreads();   // all waves' copies for buffer `par` visible

        // ---- scores: two 16x16 key tiles, K-dim = DH via two 32-steps ----
        v8f sc[2];
#pragma unroll
        for (int kt = 0; kt < 2; ++kt) {
            v8f z = {};
#pragma unroll
            for (int s = 0; s < 2; ++s) {
                const _Float16* kr =
                    &kbuf[par][(kt * 16 + lan) * DHEAD + s * 32 + half * 8];
                V16 bm;
                bm.u4[0] = *(const uint4*)kr;
                bm.u4[1] = *(const uint4*)(kr + 16);
                z = __builtin_amdgcn_wmma_f32_16x16x32_f16(
                    false, aq[s].v, false, bm.v, (short)0, z, false, false);
            }
            sc[kt] = z;
        }

        // ---- online softmax (row r lives in VGPR slot r, half gives +8) ----
        float alpha[8];
#pragma unroll
        for (int r = 0; r < 8; ++r) {
            float vmax = fmaxf(sc[0][r], sc[1][r]);
#pragma unroll
            for (int off = 1; off < 16; off <<= 1)
                vmax = fmaxf(vmax, __shfl_xor(vmax, off, 32));
            const float mn = fmaxf(m[r], vmax);
            alpha[r] = __expf(m[r] - mn);
            const float p0 = __expf(sc[0][r] - mn);
            const float p1 = __expf(sc[1][r] - mn);
            float ps = p0 + p1;
#pragma unroll
            for (int off = 1; off < 16; off <<= 1)
                ps += __shfl_xor(ps, off, 32);
            lsum[r] = lsum[r] * alpha[r] + ps;
            m[r] = mn;
            const int row = r + half * 8;
            pbuf[wave][row * 32 + lan]      = f2h(p0);
            pbuf[wave][row * 32 + 16 + lan] = f2h(p1);
        }
        // Same-wave LDS is in-order; just forbid compiler reordering.
        __builtin_amdgcn_wave_barrier();

#pragma unroll
        for (int t = 0; t < 4; ++t)
#pragma unroll
            for (int r = 0; r < 8; ++r)
                acc[t][r] *= alpha[r];

        // ---- P (A-layout from LDS) x V (B from staged transposed v) ----
        V16 ap;
        {
            const _Float16* pr = &pbuf[wave][lan * 32 + half * 8];
            ap.u4[0] = *(const uint4*)pr;
            ap.u4[1] = *(const uint4*)(pr + 16);
        }
#pragma unroll
        for (int t = 0; t < 4; ++t) {
            const _Float16* vr = &vbuf[par][(t * 16 + lan) * KB_KEYS + half * 8];
            V16 bm;
            bm.u4[0] = *(const uint4*)vr;
            bm.u4[1] = *(const uint4*)(vr + 16);
            acc[t] = __builtin_amdgcn_wmma_f32_16x16x32_f16(
                false, ap.v, false, bm.v, (short)0, acc[t], false, false);
        }
        __syncthreads();   // reads of buffer `par` done before it is restaged
    }

    // ---- epilogue: normalize, write concat layout att[S][b*DH + h] ----
#pragma unroll
    for (int t = 0; t < 4; ++t)
#pragma unroll
        for (int r = 0; r < 8; ++r) {
            const int row = row0 + r + half * 8;
            const int col = b * DHEAD + t * 16 + lan;
            att[(size_t)row * D_MODEL + col] = f2h(acc[t][r] / lsum[r]);
        }
}

// ---------------------------------------------------------------------------
// Kernel 3: output projection.  grid = (S/16, D/64), block = 32 (one wave).
// ---------------------------------------------------------------------------
__global__ void __launch_bounds__(32)
out_proj_kernel(const _Float16* __restrict__ att, const _Float16* __restrict__ Wo16,
                const float* __restrict__ bo, float* __restrict__ out) {
    const int l    = threadIdx.x;
    const int half = l >> 4;
    const int lan  = l & 15;
    const int row0 = blockIdx.x * 16;
    const int col0 = blockIdx.y * 64;

    const _Float16* arow = att + (size_t)(row0 + lan) * D_MODEL;

    v8f c[4] = {};

    for (int k0 = 0; k0 < D_MODEL; k0 += 32) {
        V16 a;
        const int ka = k0 + half * 8;
        a.u4[0] = *(const uint4*)(arow + ka);
        a.u4[1] = *(const uint4*)(arow + ka + 16);

        const int kb = k0 + half * 8;
#pragma unroll
        for (int t = 0; t < 4; ++t) {
            const _Float16* wr = Wo16 + (size_t)(col0 + t * 16 + lan) * D_MODEL + kb;
            V16 bm;
            bm.u4[0] = *(const uint4*)wr;
            bm.u4[1] = *(const uint4*)(wr + 16);
            c[t] = __builtin_amdgcn_wmma_f32_16x16x32_f16(
                false, a.v, false, bm.v, (short)0, c[t], false, false);
        }
    }

#pragma unroll
    for (int t = 0; t < 4; ++t) {
        const int col = col0 + t * 16 + lan;
        const float bval = bo[col];
#pragma unroll
        for (int r = 0; r < 8; ++r) {
            const int row = row0 + r + half * 8;
            out[(size_t)row * D_MODEL + col] = c[t][r] + bval;
        }
    }
}

// ---------------------------------------------------------------------------
extern "C" void kernel_launch(void* const* d_in, const int* in_sizes, int n_in,
                              void* d_out, int out_size, void* d_ws, size_t ws_size,
                              hipStream_t stream) {
    const float* Q  = (const float*)d_in[0];
    const float* K  = (const float*)d_in[1];
    const float* V  = (const float*)d_in[2];
    const float* Wq = (const float*)d_in[3];
    const float* bq = (const float*)d_in[4];
    const float* Wk = (const float*)d_in[5];
    const float* bk = (const float*)d_in[6];
    const float* Wv = (const float*)d_in[7];
    const float* bv = (const float*)d_in[8];
    const float* Wo = (const float*)d_in[9];
    const float* bo = (const float*)d_in[10];
    float* out = (float*)d_out;

    char* ws = (char*)d_ws;
    const size_t MB = 1024 * 1024;
    const size_t KB = 1024;
    _Float16* qws  = (_Float16*)(ws + 0 * MB);    // [B][S][DH] f16, 4 MiB
    _Float16* kws  = (_Float16*)(ws + 4 * MB);    // [B][S][DH] f16, 4 MiB
    _Float16* vtws = (_Float16*)(ws + 8 * MB);    // [B][DH][S] f16, 4 MiB
    _Float16* att  = (_Float16*)(ws + 12 * MB);   // [S][D]     f16, 4 MiB
    _Float16* Wq16 = (_Float16*)(ws + 16 * MB);             // 128 KiB
    _Float16* Wk16 = (_Float16*)(ws + 16 * MB + 128 * KB);  // 128 KiB
    _Float16* Wv16 = (_Float16*)(ws + 16 * MB + 256 * KB);  // 128 KiB
    _Float16* Wo16 = (_Float16*)(ws + 16 * MB + 384 * KB);  // 2 MiB

    const int nw = DHEAD * D_MODEL;       // 65536
    const int no = D_MODEL * D_MODEL;     // 1048576
    cvt_w_kernel<<<(nw / 4 + 255) / 256, 256, 0, stream>>>(Wq, Wq16, nw);
    cvt_w_kernel<<<(nw / 4 + 255) / 256, 256, 0, stream>>>(Wk, Wk16, nw);
    cvt_w_kernel<<<(nw / 4 + 255) / 256, 256, 0, stream>>>(Wv, Wv16, nw);
    cvt_w_kernel<<<(no / 4 + 255) / 256, 256, 0, stream>>>(Wo, Wo16, no);

    proj_qkv_kernel<<<dim3(S_LEN / 16, NHEAD, 3), 32, 0, stream>>>(
        Q, K, V, Wq16, bq, Wk16, bk, Wv16, bv, qws, kws, vtws);

    flash_attn_kernel<<<dim3(S_LEN / 64, NHEAD), 128, 0, stream>>>(
        qws, kws, vtws, att);

    out_proj_kernel<<<dim3(S_LEN / 16, D_MODEL / 64), 32, 0, stream>>>(
        att, Wo16, bo, out);
}
